// GPT_25615184954011
// MI455X (gfx1250) — compile-verified, matrix operand order checked by reading, not run on hardware
//
#include <hip/hip_runtime.h>
#include <hip/hip_bf16.h>
#include <math.h>

// ---------------- model dims ----------------
#define BB 2
#define TT 1024
#define LL 4
#define HH 12
#define DD 768
#define DHH 64
#define FFF 3072
#define VV 50257

typedef _Float16 half_t;
typedef __attribute__((ext_vector_type(16))) _Float16 v16h;
typedef __attribute__((ext_vector_type(8)))  _Float16 v8h;
typedef __attribute__((ext_vector_type(8)))  float    v8f;
typedef __attribute__((ext_vector_type(4)))  int      v4i;

#define APAD 40   // LDS row stride in halves (80B, 16B aligned, conflict-spreading)

#define AS_GLOBAL __attribute__((address_space(1)))
#define AS_LOCAL  __attribute__((address_space(3)))

// gfx1250 async global->LDS staging path (ASYNCcnt-tracked), with fallback.
#if defined(__has_builtin)
#if __has_builtin(__builtin_amdgcn_global_load_async_to_lds_b128)
#define USE_ASYNC 1
#endif
#endif
#ifndef USE_ASYNC
#define USE_ASYNC 0
#endif

__device__ __forceinline__ void async_wait_le2() {
#if USE_ASYNC
  asm volatile("s_wait_asynccnt 0x2" ::: "memory");
#endif
}
__device__ __forceinline__ void async_wait_le0() {
#if USE_ASYNC
  asm volatile("s_wait_asynccnt 0x0" ::: "memory");
#endif
}

// ------------------------------------------------------------------
// block reductions (wave32, 256-thread blocks = 8 waves)
// ------------------------------------------------------------------
__device__ __forceinline__ float warpSum(float v) {
#pragma unroll
  for (int o = 16; o > 0; o >>= 1) v += __shfl_down(v, o, 32);
  return v;
}
__device__ __forceinline__ float warpMax(float v) {
#pragma unroll
  for (int o = 16; o > 0; o >>= 1) v = fmaxf(v, __shfl_down(v, o, 32));
  return v;
}
__device__ __forceinline__ float blockSum(float v, float* sm) {
  int lane = threadIdx.x & 31, w = threadIdx.x >> 5;
  v = warpSum(v);
  if (lane == 0) sm[w] = v;
  __syncthreads();
  float r = 0.f;
#pragma unroll
  for (int i = 0; i < 8; ++i) r += sm[i];
  __syncthreads();
  return r;
}
__device__ __forceinline__ float blockMax(float v, float* sm) {
  int lane = threadIdx.x & 31, w = threadIdx.x >> 5;
  v = warpMax(v);
  if (lane == 0) sm[w] = v;
  __syncthreads();
  float r = -3.4e38f;
#pragma unroll
  for (int i = 0; i < 8; ++i) r = fmaxf(r, sm[i]);
  __syncthreads();
  return r;
}

// ------------------------------------------------------------------
// Generic WMMA GEMM:  C[z](M,N) = A[z](M,K) * Bt[z](N,K)^T  (+bias,+res)
//   mode 0: Cf[m*ldc+n] = acc + bias + residual          (fp32 out)
//   mode 1: Ch[m*ldc+n] = f16(acc + bias), optional relu (f16 out)
//   mode 2: Ch[n*ldc+m] = f16(acc + bias)                (f16 transposed out)
// z decomposed as z1=z/zdiv, z2=z%zdiv with per-operand strides.
// Double-buffered LDS staging via GLOBAL_LOAD_ASYNC_TO_LDS_B128 (ASYNCcnt);
// out-of-range staging rows are clamped (never stored: epilogue masks them),
// which keeps EXEC all-1s through the WMMA region.
// ------------------------------------------------------------------
__global__ __launch_bounds__(256)
void wmma_gemm(const half_t* __restrict__ A, long lda, long sA1, long sA2,
               const half_t* __restrict__ Bt, long ldb, long sB1, long sB2,
               float* Cf, half_t* Ch, long ldc, long sC1, long sC2,
               const float* __restrict__ bias,
               const float* __restrict__ residual,
               int M, int N, int K, int zdiv, int mode, int relu)
{
  __shared__ __align__(16) half_t la[2][64 * APAD];
  __shared__ __align__(16) half_t lb[2][64 * APAD];

  const int z  = blockIdx.z;
  const int z1 = z / zdiv, z2 = z % zdiv;
  A  += (long)z1 * sA1 + (long)z2 * sA2;
  Bt += (long)z1 * sB1 + (long)z2 * sB2;
  const long coff = (long)z1 * sC1 + (long)z2 * sC2;
  if (Cf) Cf += coff;
  if (Ch) Ch += coff;
  if (residual) residual += coff;

  const int m0 = blockIdx.x * 64;
  const int n0 = blockIdx.y * 64;
  const int t    = threadIdx.x;
  const int lane = t & 31;
  const int wave = t >> 5;           // 0..7
  const int wm   = wave & 3;         // m-subtile 0..3
  const int wn   = (wave >> 2) * 2;  // n-subtiles wn, wn+1

  const int srow = t >> 2;           // 0..63  (staging row)
  const int scol = (t & 3) * 8;      // 0,8,16,24 halves (16B chunk)

  // clamped staging rows (OOB rows read valid memory; epilogue masks them)
  const int cm = (m0 + srow < M) ? (m0 + srow) : (M - 1);
  const int cn = (n0 + srow < N) ? (n0 + srow) : (N - 1);
  const half_t* gA = A  + (long)cm * lda + scol;
  const half_t* gB = Bt + (long)cn * ldb + scol;

  auto stage = [&](int kk, int bufi) {
#if USE_ASYNC
    __builtin_amdgcn_global_load_async_to_lds_b128(
        (AS_GLOBAL v4i*)(gA + kk),
        (AS_LOCAL v4i*)&la[bufi][srow * APAD + scol], 0, 0);
    __builtin_amdgcn_global_load_async_to_lds_b128(
        (AS_GLOBAL v4i*)(gB + kk),
        (AS_LOCAL v4i*)&lb[bufi][srow * APAD + scol], 0, 0);
#else
    uint4 va = *(const uint4*)(gA + kk);
    uint4 vb = *(const uint4*)(gB + kk);
    *(uint4*)(&la[bufi][srow * APAD + scol]) = va;
    *(uint4*)(&lb[bufi][srow * APAD + scol]) = vb;
#endif
  };

  v8f acc0 = {}; v8f acc1 = {};

  const int r  = lane & 15;
  const int hk = (lane >> 4) * 8;    // K sub-offset per documented 16-bit layout

  stage(0, 0);                        // prologue: tile 0 -> buffer 0
  int buf = 0;
  for (int k0 = 0; k0 < K; k0 += 32) {
    const int nxt = k0 + 32;
    if (nxt < K) {                    // issue next tile into the other buffer
      stage(nxt, buf ^ 1);
      async_wait_le2();               // drain current tile's 2 async loads
    } else {
      async_wait_le0();
    }
    __syncthreads();

    // ---- load fragments (16x32 A, 32x16 B) per ISA VGPR layout ----
    const half_t* pa  = &la[buf][(wm * 16 + r) * APAD + hk];
    const half_t* pb0 = &lb[buf][((wn + 0) * 16 + r) * APAD + hk];
    const half_t* pb1 = &lb[buf][((wn + 1) * 16 + r) * APAD + hk];
    v8h alo = *(const v8h*)(pa);       v8h ahi = *(const v8h*)(pa + 16);
    v8h b0l = *(const v8h*)(pb0);      v8h b0h = *(const v8h*)(pb0 + 16);
    v8h b1l = *(const v8h*)(pb1);      v8h b1h = *(const v8h*)(pb1 + 16);
    v16h af, bf0, bf1;
#pragma unroll
    for (int i = 0; i < 8; ++i) {
      af[i]  = alo[i]; af[i + 8]  = ahi[i];
      bf0[i] = b0l[i]; bf0[i + 8] = b0h[i];
      bf1[i] = b1l[i]; bf1[i + 8] = b1h[i];
    }
    acc0 = __builtin_amdgcn_wmma_f32_16x16x32_f16(false, af, false, bf0,
                                                  (short)0, acc0, false, false);
    acc1 = __builtin_amdgcn_wmma_f32_16x16x32_f16(false, af, false, bf1,
                                                  (short)0, acc1, false, false);
    buf ^= 1;
    __syncthreads();                  // protect buffer being restaged next iter
  }

  // ---- epilogue (C layout: VGPR r -> row r / r+8; lane&15 -> col) ----
  const int cl = lane & 15;
  const int rh = (lane >> 4) * 8;
#pragma unroll
  for (int s2 = 0; s2 < 2; ++s2) {
    const v8f acc = s2 ? acc1 : acc0;
    const int nn = n0 + (wn + s2) * 16 + cl;
    if (nn >= N) continue;
    const float bb = bias ? bias[nn] : 0.f;
#pragma unroll
    for (int rr = 0; rr < 8; ++rr) {
      const int mm = m0 + wm * 16 + rh + rr;
      if (mm >= M) continue;
      float vv = acc[rr] + bb;
      if (mode == 0) {
        if (residual) vv += residual[(long)mm * ldc + nn];
        Cf[(long)mm * ldc + nn] = vv;
      } else if (mode == 1) {
        if (relu && vv < 0.f) vv = 0.f;
        Ch[(long)mm * ldc + nn] = (half_t)vv;
      } else {
        Ch[(long)nn * ldc + mm] = (half_t)vv;
      }
    }
  }
}

// ------------------------------------------------------------------
// embedding + sinusoidal PE : x[row][d] = emb[tok[row]][d] + pe(t,d)
// ------------------------------------------------------------------
__global__ __launch_bounds__(256)
void embed_kernel(const int* __restrict__ tokens, const float* __restrict__ emb,
                  float* __restrict__ x)
{
  const int row = blockIdx.x;            // 0..B*T-1
  const int tok = tokens[row];
  const float pos = (float)(row % TT);
  const float c = -logf(10000.f) / (float)DD;
  for (int d = threadIdx.x; d < DD; d += blockDim.x) {
    const int i2 = (d >> 1) * 2;
    const float div = __expf((float)i2 * c);
    const float pe = (d & 1) ? __cosf(pos * div) : __sinf(pos * div);
    x[(long)row * DD + d] = emb[(long)tok * DD + d] + pe;
  }
}

// ------------------------------------------------------------------
// layernorm -> f16
// ------------------------------------------------------------------
__global__ __launch_bounds__(256)
void layernorm_f16(const float* __restrict__ x, const float* __restrict__ g,
                   const float* __restrict__ b, half_t* __restrict__ y)
{
  __shared__ float sm[8];
  const long row = blockIdx.x;
  const float* xr = x + row * DD;
  float s = 0.f;
  for (int i = threadIdx.x; i < DD; i += blockDim.x) s += xr[i];
  const float mu = blockSum(s, sm) / (float)DD;
  float v = 0.f;
  for (int i = threadIdx.x; i < DD; i += blockDim.x) { float d = xr[i] - mu; v += d * d; }
  const float inv = rsqrtf(blockSum(v, sm) / (float)DD + 1e-5f);
  for (int i = threadIdx.x; i < DD; i += blockDim.x)
    y[row * DD + i] = (half_t)((xr[i] - mu) * inv * g[i] + b[i]);
}

// ------------------------------------------------------------------
// rmsnorm -> f16
// ------------------------------------------------------------------
__global__ __launch_bounds__(256)
void rmsnorm_f16(const float* __restrict__ x, const float* __restrict__ w,
                 half_t* __restrict__ y)
{
  __shared__ float sm[8];
  const long row = blockIdx.x;
  const float* xr = x + row * DD;
  float s = 0.f;
  for (int i = threadIdx.x; i < DD; i += blockDim.x) s += xr[i] * xr[i];
  const float inv = rsqrtf(blockSum(s, sm) / (float)DD + 1e-6f);
  for (int i = threadIdx.x; i < DD; i += blockDim.x)
    y[row * DD + i] = (half_t)(xr[i] * inv * w[i]);
}

// ------------------------------------------------------------------
// causal scaled softmax row: S fp32 -> P f16
// ------------------------------------------------------------------
__global__ __launch_bounds__(256)
void softmax_causal(const float* __restrict__ S, half_t* __restrict__ P, float scale)
{
  __shared__ float sm[8];
  const int t  = blockIdx.x;
  const int bh = blockIdx.y;
  const float* srow = S + ((long)bh * TT + t) * TT;
  half_t*      prow = P + ((long)bh * TT + t) * TT;
  const int len = t + 1;
  float mx = -3.4e38f;
  for (int i = threadIdx.x; i < len; i += blockDim.x) mx = fmaxf(mx, srow[i] * scale);
  mx = blockMax(mx, sm);
  float sum = 0.f;
  for (int i = threadIdx.x; i < len; i += blockDim.x) sum += __expf(srow[i] * scale - mx);
  sum = blockSum(sum, sm);
  const float inv = 1.f / sum;
  for (int i = threadIdx.x; i < TT; i += blockDim.x) {
    float p = (i < len) ? __expf(srow[i] * scale - mx) * inv : 0.f;
    prow[i] = (half_t)p;
  }
}

// ------------------------------------------------------------------
// plain transpose fp32 [K][N] -> f16 [N][K]
// ------------------------------------------------------------------
__global__ __launch_bounds__(256)
void transpose_to_f16(const float* __restrict__ in, half_t* __restrict__ out,
                      int K, int N)
{
  const long total = (long)K * N;
  for (long o = (long)blockIdx.x * blockDim.x + threadIdx.x; o < total;
       o += (long)gridDim.x * blockDim.x) {
    const long n = o / K, k = o % K;
    out[o] = (half_t)in[k * (long)N + n];
  }
}

// ------------------------------------------------------------------
// QKV weight: fp32 [H][D][DH] -> f16 [H*DH][D]
// ------------------------------------------------------------------
__global__ __launch_bounds__(256)
void qkv_transpose_f16(const float* __restrict__ in, half_t* __restrict__ out)
{
  const long total = (long)HH * DD * DHH;
  for (long o = (long)blockIdx.x * blockDim.x + threadIdx.x; o < total;
       o += (long)gridDim.x * blockDim.x) {
    const long d  = o % DD;
    const long he = o / DD;
    const long h  = he / DHH, e = he % DHH;
    out[o] = (half_t)in[(h * DD + d) * DHH + e];
  }
}

// ------------------------------------------------------------------
// host-side helpers
// ------------------------------------------------------------------
static void gemm(hipStream_t st,
                 const half_t* A, long lda, long sA1, long sA2,
                 const half_t* Bt, long ldb, long sB1, long sB2,
                 float* Cf, half_t* Ch, long ldc, long sC1, long sC2,
                 const float* bias, const float* residual,
                 int M, int N, int K, int nz, int zdiv, int mode, int relu)
{
  dim3 g((M + 63) / 64, (N + 63) / 64, nz);
  wmma_gemm<<<g, 256, 0, st>>>(A, lda, sA1, sA2, Bt, ldb, sB1, sB2,
                               Cf, Ch, ldc, sC1, sC2, bias, residual,
                               M, N, K, zdiv, mode, relu);
}

extern "C" void kernel_launch(void* const* d_in, const int* in_sizes, int n_in,
                              void* d_out, int out_size, void* d_ws, size_t ws_size,
                              hipStream_t stream)
{
  (void)in_sizes; (void)n_in; (void)out_size; (void)ws_size;
  // ---- inputs (setup_inputs order) ----
  const int*   tokens = (const int*)  d_in[0];
  const float* emb    = (const float*)d_in[1];
  const float* wq     = (const float*)d_in[2];
  const float* bq     = (const float*)d_in[3];
  const float* wk     = (const float*)d_in[4];
  const float* bk     = (const float*)d_in[5];
  const float* wv     = (const float*)d_in[6];
  const float* bv     = (const float*)d_in[7];
  const float* wo     = (const float*)d_in[8];
  const float* bo     = (const float*)d_in[9];
  const float* ln1_g  = (const float*)d_in[10];
  const float* ln1_b  = (const float*)d_in[11];
  const float* ln2_g  = (const float*)d_in[12];
  const float* ln2_b  = (const float*)d_in[13];
  const float* w1     = (const float*)d_in[14];
  const float* b1     = (const float*)d_in[15];
  const float* w2     = (const float*)d_in[16];
  const float* b2     = (const float*)d_in[17];
  const float* rms_w  = (const float*)d_in[18];
  const float* wu     = (const float*)d_in[19];
  const float* bu     = (const float*)d_in[20];
  float* logits = (float*)d_out;

  // ---- workspace layout (bytes, 256B aligned) ----
  char* ws = (char*)d_ws;
  size_t off = 0;
  auto take = [&](size_t bytes) { char* p = ws + off; off = (off + bytes + 255) & ~(size_t)255; return p; };
  const long MT = (long)BB * TT;                 // 2048 rows
  float*  xf  = (float*) take(MT * DD * 4);      // residual stream fp32
  half_t* xn  = (half_t*)take(MT * DD * 2);      // normed activations f16
  half_t* qh  = (half_t*)take(MT * DD * 2);      // q [b][t][h*64+e]
  half_t* kh  = (half_t*)take(MT * DD * 2);      // k [b][t][h*64+e]
  half_t* vT  = (half_t*)take(MT * DD * 2);      // v [b][h][e][t]
  half_t* att = (half_t*)take(MT * DD * 2);      // attn out [b][t][h*64+e]
  half_t* hff = (half_t*)take(MT * FFF * 2);     // ffn hidden f16
  half_t* WqT = (half_t*)take((long)DD * DD * 2);
  half_t* WkT = (half_t*)take((long)DD * DD * 2);
  half_t* WvT = (half_t*)take((long)DD * DD * 2);
  half_t* WoT = (half_t*)take((long)DD * DD * 2);
  half_t* W1T = (half_t*)take((long)FFF * DD * 2);
  half_t* W2T = (half_t*)take((long)DD * FFF * 2);
  float*  S   = (float*) take((long)BB * HH * TT * TT * 4);   // scores fp32
  half_t* P   = (half_t*)take((long)BB * HH * TT * TT * 2);   // probs f16
  half_t* wuT = (half_t*)S;   // reuse scores region after the layer loop (77MB < 100MB)

  const float scale = 1.f / sqrtf((float)DHH);

  // ---- embedding + positional encoding ----
  embed_kernel<<<dim3(MT), 256, 0, stream>>>(tokens, emb, xf);

  for (int l = 0; l < LL; ++l) {
    // weight conversion (f16, [N][K])
    {
      long n;
      n = (long)HH * DD * DHH;
      qkv_transpose_f16<<<dim3((n + 255) / 256), 256, 0, stream>>>(wq + l * n, WqT);
      qkv_transpose_f16<<<dim3((n + 255) / 256), 256, 0, stream>>>(wk + l * n, WkT);
      qkv_transpose_f16<<<dim3((n + 255) / 256), 256, 0, stream>>>(wv + l * n, WvT);
      n = (long)DD * DD;
      transpose_to_f16<<<dim3((n + 255) / 256), 256, 0, stream>>>(wo + l * n, WoT, DD, DD);
      n = (long)DD * FFF;
      transpose_to_f16<<<dim3((n + 255) / 256), 256, 0, stream>>>(w1 + l * n, W1T, DD, FFF);
      transpose_to_f16<<<dim3((n + 255) / 256), 256, 0, stream>>>(w2 + l * n, W2T, FFF, DD);
    }

    // ln1
    layernorm_f16<<<dim3(MT), 256, 0, stream>>>(xf, ln1_g + l * DD, ln1_b + l * DD, xn);

    // q, k projections: [2048x768] x [768x768]^T -> f16
    gemm(stream, xn, DD, 0, 0, WqT, DD, 0, 0, nullptr, qh, DD, 0, 0,
         bq + (long)l * HH * DHH, nullptr, (int)MT, DD, DD, 1, 1, 1, 0);
    gemm(stream, xn, DD, 0, 0, WkT, DD, 0, 0, nullptr, kh, DD, 0, 0,
         bk + (long)l * HH * DHH, nullptr, (int)MT, DD, DD, 1, 1, 1, 0);
    // v projection, transposed store per batch b: vT[b][h*64+e][t]
    gemm(stream, xn, DD, (long)TT * DD, 0, WvT, DD, 0, 0, nullptr, vT, TT,
         (long)HH * DHH * TT, 0, bv + (long)l * HH * DHH, nullptr,
         TT, DD, DD, BB, 1, 2, 0);

    // scores: S[b,h] = q[b,h] (1024x64) x k[b,h]^T -> fp32
    gemm(stream, qh, DD, (long)TT * DD, DHH, kh, DD, (long)TT * DD, DHH,
         S, nullptr, TT, (long)HH * TT * TT, (long)TT * TT, nullptr, nullptr,
         TT, TT, DHH, BB * HH, HH, 0, 0);

    // causal softmax with scale -> P f16
    softmax_causal<<<dim3(TT, BB * HH), 256, 0, stream>>>(S, P, scale);

    // attn out: att[b,h] = P[b,h] (1024x1024) x vT[b,h] (64x1024)^T -> f16 concat-head
    gemm(stream, P, TT, (long)HH * TT * TT, (long)TT * TT,
         vT, TT, (long)HH * DHH * TT, (long)DHH * TT,
         nullptr, att, DD, (long)TT * DD, DHH, nullptr, nullptr,
         TT, DHH, TT, BB * HH, HH, 1, 0);

    // x += att @ wo + bo  (fp32 residual update)
    gemm(stream, att, DD, 0, 0, WoT, DD, 0, 0, xf, nullptr, DD, 0, 0,
         bo + (long)l * DD, xf, (int)MT, DD, DD, 1, 1, 0, 0);

    // ln2
    layernorm_f16<<<dim3(MT), 256, 0, stream>>>(xf, ln2_g + l * DD, ln2_b + l * DD, xn);

    // h = relu(xn @ w1 + b1) -> f16
    gemm(stream, xn, DD, 0, 0, W1T, DD, 0, 0, nullptr, hff, FFF, 0, 0,
         b1 + (long)l * FFF, nullptr, (int)MT, FFF, DD, 1, 1, 1, 1);

    // x += h @ w2 + b2
    gemm(stream, hff, FFF, 0, 0, W2T, FFF, 0, 0, xf, nullptr, DD, 0, 0,
         b2 + (long)l * DD, xf, (int)MT, DD, FFF, 1, 1, 0, 0);
  }

  // final rmsnorm -> f16
  rmsnorm_f16<<<dim3(MT), 256, 0, stream>>>(xf, rms_w, xn);

  // wu fp32 [768][50257] -> f16 [50257][768] (reuses scores region)
  {
    long n = (long)DD * VV;
    transpose_to_f16<<<dim3((n + 255) / 256), 256, 0, stream>>>(wu, wuT, DD, VV);
  }

  // logits = xn @ wu + bu   [2048 x 50257] fp32
  gemm(stream, xn, DD, 0, 0, wuT, DD, 0, 0, logits, nullptr, VV, 0, 0,
       bu, nullptr, (int)MT, VV, DD, 1, 1, 0, 0);
}
